// AdaptiveSparseAttention_38706245272122
// MI455X (gfx1250) — matrix operand order repaired and checked by reading, not compiled, not run on hardware
//
#include <hip/hip_runtime.h>
#include <hip/hip_bf16.h>
#include <math.h>
#include <stdint.h>

typedef _Float16 f16;
typedef __attribute__((ext_vector_type(16))) _Float16 v16h;
typedef __attribute__((ext_vector_type(8)))  float    v8f;

#define B_    4
#define L_    1024
#define D_    1024
#define H_    16
#define HD_   64
#define KTOP  716        // int(1024 * 0.7)
#define HALFW 16
#define THRESH 0.1f
#define SCALE_ 0.125f    // 64^-0.5

// ---------------------------------------------------------------------------
// WMMA fragment helpers (CDNA5 wave32, 16x16x32 f16 -> f32)
// A (16x32, MxK): lanes 0-15 row M=lane, K in {0..15}; lanes 16-31 same
// row, K in {16..31}. B (32x16, KxN): lane n=lane&15; lanes 0-15 hold
// K=0..15 (2 per vgpr), lanes 16-31 hold K=16..31.
// ---------------------------------------------------------------------------
__device__ __forceinline__ v16h load_a_frag(const f16* a, int lda) {
  int lane = threadIdx.x & 31;
  int m = lane & 15;
  int kb = (lane & 16) ? 16 : 0;
  const f16* p = a + m * lda + kb;
  v16h r;
#pragma unroll
  for (int t = 0; t < 8; ++t) r[t] = p[t];
#pragma unroll
  for (int t = 0; t < 8; ++t) r[t + 8] = p[8 + t];
  return r;
}

// B element (k,n) at b[k*ldb + n]  (K-major storage)
__device__ __forceinline__ v16h load_b_frag_k(const f16* b, int ldb) {
  int lane = threadIdx.x & 31;
  int n = lane & 15;
  int kb = (lane & 16) ? 16 : 0;
  v16h r;
#pragma unroll
  for (int t = 0; t < 16; ++t) r[t] = b[(kb + t) * ldb + n];
  return r;
}

// B element (k,n) at b[n*ldb + k]  (N-major storage, i.e. B = W row-major)
__device__ __forceinline__ v16h load_b_frag_n(const f16* b, int ldb) {
  int lane = threadIdx.x & 31;
  int n = lane & 15;
  int kb = (lane & 16) ? 16 : 0;
  const f16* p = b + n * ldb + kb;
  v16h r;
#pragma unroll
  for (int t = 0; t < 16; ++t) r[t] = p[t];
  return r;
}

__device__ __forceinline__ v8f wmma16(v16h a, v16h b, v8f c) {
  return __builtin_amdgcn_wmma_f32_16x16x32_f16(false, a, false, b, (short)0, c,
                                                false, false);
}

__device__ __forceinline__ float gelu_f(float x) {
  return 0.5f * x * (1.0f + erff(x * 0.7071067811865476f));
}

// ---------------------------------------------------------------------------
// f32 -> f16 conversion
// ---------------------------------------------------------------------------
__global__ void k_cvt(const float* __restrict__ in, f16* __restrict__ out, int n) {
  int i = blockIdx.x * blockDim.x + threadIdx.x;
  if (i < n) out[i] = (f16)in[i];
}

// ---------------------------------------------------------------------------
// QKV GEMM: [4096 x 1024] x [1024 x 3072] -> scatter into q/k/v [B,H,L,HD] f16
// One wave computes a 16x64 tile.
// ---------------------------------------------------------------------------
__global__ void k_gemm_qkv(const f16* __restrict__ xh, const f16* __restrict__ wh,
                           f16* __restrict__ qh, f16* __restrict__ kh,
                           f16* __restrict__ vh) {
  const int NT = 48;  // 3072 / 64
  int wave = blockIdx.x * (blockDim.x >> 5) + (threadIdx.x >> 5);
  int mt = wave / NT, nt = wave % NT;
  int m0 = mt * 16, n0 = nt * 64;
  v8f a0 = {}, a1 = {}, a2 = {}, a3 = {};
  for (int k0 = 0; k0 < D_; k0 += 32) {
    __builtin_prefetch(xh + m0 * D_ + k0 + 64, 0, 1);
    v16h a  = load_a_frag(xh + m0 * D_ + k0, D_);
    v16h b0 = load_b_frag_n(wh + (n0 +  0) * D_ + k0, D_);
    v16h b1 = load_b_frag_n(wh + (n0 + 16) * D_ + k0, D_);
    v16h b2 = load_b_frag_n(wh + (n0 + 32) * D_ + k0, D_);
    v16h b3 = load_b_frag_n(wh + (n0 + 48) * D_ + k0, D_);
    a0 = wmma16(a, b0, a0);
    a1 = wmma16(a, b1, a1);
    a2 = wmma16(a, b2, a2);
    a3 = wmma16(a, b3, a3);
  }
  int lane = threadIdx.x & 31;
  int n = lane & 15, mb = (lane & 16) ? 8 : 0;
  int which = n0 / 1024;
  int h = (n0 % 1024) >> 6;
  int bb = m0 / 1024;
  f16* dst = (which == 0) ? qh : (which == 1) ? kh : vh;
  f16* base = dst + (size_t)((bb * H_ + h) * L_) * HD_;
#pragma unroll
  for (int v = 0; v < 8; ++v) {
    int i = (m0 & 1023) + mb + v;
    f16* row = base + (size_t)i * HD_;
    row[ 0 + n] = (f16)a0[v];
    row[16 + n] = (f16)a1[v];
    row[32 + n] = (f16)a2[v];
    row[48 + n] = (f16)a3[v];
  }
}

// ---------------------------------------------------------------------------
// scores[b,h] = (q @ k^T) * SCALE    (one wave -> 16x64 tile)
// ---------------------------------------------------------------------------
__global__ void k_scores(const f16* __restrict__ qh, const f16* __restrict__ kh,
                         float* __restrict__ sc) {
  int bh = blockIdx.y;
  int wave = blockIdx.x * 8 + (threadIdx.x >> 5);  // 0..1023
  int mt = wave >> 4, nt = wave & 15;
  int m0 = mt * 16, n0 = nt * 64;
  const f16* qb = qh + (size_t)bh * L_ * HD_;
  const f16* kb = kh + (size_t)bh * L_ * HD_;
  v8f a0 = {}, a1 = {}, a2 = {}, a3 = {};
#pragma unroll
  for (int k0 = 0; k0 < HD_; k0 += 32) {
    v16h a  = load_a_frag(qb + m0 * HD_ + k0, HD_);
    v16h b0 = load_b_frag_n(kb + (n0 +  0) * HD_ + k0, HD_);
    v16h b1 = load_b_frag_n(kb + (n0 + 16) * HD_ + k0, HD_);
    v16h b2 = load_b_frag_n(kb + (n0 + 32) * HD_ + k0, HD_);
    v16h b3 = load_b_frag_n(kb + (n0 + 48) * HD_ + k0, HD_);
    a0 = wmma16(a, b0, a0);
    a1 = wmma16(a, b1, a1);
    a2 = wmma16(a, b2, a2);
    a3 = wmma16(a, b3, a3);
  }
  float* out = sc + (size_t)bh * L_ * L_;
  int lane = threadIdx.x & 31;
  int n = lane & 15, mb = (lane & 16) ? 8 : 0;
#pragma unroll
  for (int v = 0; v < 8; ++v) {
    float* row = out + (size_t)(m0 + mb + v) * L_ + n0;
    row[ 0 + n] = a0[v] * SCALE_;
    row[16 + n] = a1[v] * SCALE_;
    row[32 + n] = a2[v] * SCALE_;
    row[48 + n] = a3[v] * SCALE_;
  }
}

// ---------------------------------------------------------------------------
// Per-row K_TOP-th largest score, via exact radix binary search on
// order-preserving uint keys. One wave per row; 32 f32 values per lane held
// in registers; wave32 shuffle reductions.
// ---------------------------------------------------------------------------
__global__ void k_thresh(const float* __restrict__ sc, float* __restrict__ thr) {
  int wave = blockIdx.x * 8 + (threadIdx.x >> 5);  // global row id (bh*L + i)
  const float* row = sc + (size_t)wave * L_;
  int lane = threadIdx.x & 31;
  unsigned key[32];
#pragma unroll
  for (int t = 0; t < 32; ++t) {
    unsigned b = __float_as_uint(row[lane + 32 * t]);
    key[t] = (b & 0x80000000u) ? ~b : (b | 0x80000000u);
  }
  unsigned t = 0u;
  for (int bit = 31; bit >= 0; --bit) {
    unsigned cand = t | (1u << bit);
    int c = 0;
#pragma unroll
    for (int q = 0; q < 32; ++q) c += (key[q] >= cand) ? 1 : 0;
#pragma unroll
    for (int m = 16; m; m >>= 1) c += __shfl_xor(c, m, 32);
    if (c >= KTOP) t = cand;
  }
  if (lane == 0) {
    unsigned b = (t & 0x80000000u) ? (t & 0x7fffffffu) : ~t;
    thr[wave] = __uint_as_float(b);
  }
}

// ---------------------------------------------------------------------------
// pooled = (mean + max)/2 over L
// ---------------------------------------------------------------------------
__global__ void k_pool(const float* __restrict__ x, float* __restrict__ pooled) {
  int idx = blockIdx.x * blockDim.x + threadIdx.x;  // b*D + d
  if (idx >= B_ * D_) return;
  int b = idx / D_, d = idx % D_;
  const float* p = x + (size_t)b * L_ * D_ + d;
  float s = 0.f, m = -__builtin_inff();
  for (int l = 0; l < L_; ++l) {
    float v = p[(size_t)l * D_];
    s += v;
    if (v > m) m = v;
  }
  pooled[idx] = (s * (1.0f / L_) + m) * 0.5f;
}

// ---------------------------------------------------------------------------
// Wave-per-output dot:  out[b*OUT + j] = dot(in[b,:], w[j,:]) + bias[j]
// mode==1 applies exact gelu.
// ---------------------------------------------------------------------------
__global__ void k_dot(const float* __restrict__ in, const float* __restrict__ w,
                      const float* __restrict__ bias, float* __restrict__ out,
                      int K, int OUT, int mode) {
  int wave = blockIdx.x * (blockDim.x >> 5) + (threadIdx.x >> 5);
  int b = wave / OUT, j = wave % OUT;
  const float* a  = in + (size_t)b * K;
  const float* ww = w + (size_t)j * K;
  int lane = threadIdx.x & 31;
  float s = 0.f;
  for (int k = lane; k < K; k += 32) s += a[k] * ww[k];
#pragma unroll
  for (int m = 16; m; m >>= 1) s += __shfl_xor(s, m, 32);
  if (lane == 0) {
    s += bias[j];
    if (mode) s = gelu_f(s);
    out[wave] = s;
  }
}

// ---------------------------------------------------------------------------
// LayerNorm over D per batch + exact gelu, in place on h1 [B,D]
// ---------------------------------------------------------------------------
__global__ void k_ln_gelu(float* __restrict__ h, const float* __restrict__ g,
                          const float* __restrict__ bta) {
  __shared__ float red[256];
  int b = blockIdx.x, t = threadIdx.x;
  float* row = h + (size_t)b * D_;
  float s = 0.f;
  for (int k = t; k < D_; k += 256) s += row[k];
  red[t] = s;
  __syncthreads();
  for (int off = 128; off; off >>= 1) {
    if (t < off) red[t] += red[t + off];
    __syncthreads();
  }
  float mean = red[0] * (1.0f / D_);
  __syncthreads();
  float vs = 0.f;
  for (int k = t; k < D_; k += 256) {
    float d = row[k] - mean;
    vs += d * d;
  }
  red[t] = vs;
  __syncthreads();
  for (int off = 128; off; off >>= 1) {
    if (t < off) red[t] += red[t + off];
    __syncthreads();
  }
  float inv = rsqrtf(red[0] * (1.0f / D_) + 1e-5f);
  for (int k = t; k < D_; k += 256) {
    float v = (row[k] - mean) * inv * g[k] + bta[k];
    row[k] = gelu_f(v);
  }
}

// ---------------------------------------------------------------------------
// logits = h2 @ w3^T + b3 + pattern_bias; pw = softmax(logits). Block of 96
// per batch (3 waves, one per class).
// ---------------------------------------------------------------------------
__global__ void k_logits(const float* __restrict__ h2, const float* __restrict__ w3,
                         const float* __restrict__ b3, const float* __restrict__ pbias,
                         float* __restrict__ pw) {
  __shared__ float lg[3];
  int b = blockIdx.x, t = threadIdx.x;
  int wv = t >> 5, lane = t & 31;
  const float* a  = h2 + (size_t)b * 512;
  const float* ww = w3 + (size_t)wv * 512;
  float s = 0.f;
  for (int k = lane; k < 512; k += 32) s += a[k] * ww[k];
#pragma unroll
  for (int m = 16; m; m >>= 1) s += __shfl_xor(s, m, 32);
  if (lane == 0) lg[wv] = s + b3[wv] + pbias[wv];
  __syncthreads();
  if (t == 0) {
    float mx = fmaxf(lg[0], fmaxf(lg[1], lg[2]));
    float e0 = expf(lg[0] - mx), e1 = expf(lg[1] - mx), e2 = expf(lg[2] - mx);
    float inv = 1.0f / (e0 + e1 + e2);
    pw[b * 3 + 0] = e0 * inv;
    pw[b * 3 + 1] = e1 * inv;
    pw[b * 3 + 2] = e2 * inv;
  }
}

// ---------------------------------------------------------------------------
// Fused mask + softmax + (P @ V). Block = 128 threads per (bh, 16-row tile).
// Probabilities staged in LDS as f16; V chunks staged in LDS via CDNA5
// async global->LDS copies (ASYNCcnt path); 4 parallel WMMA K-loops.
// ---------------------------------------------------------------------------
__global__ void k_attn(const float* __restrict__ sc, const float* __restrict__ thr,
                       const float* __restrict__ pw, const f16* __restrict__ vh,
                       f16* __restrict__ ao) {
  __shared__ f16 ph[16][1024];           // 32 KB probability tile
  __shared__ f16 vtile[128 * 64];        // 16 KB staged V chunk
  __shared__ float partf[128];
  __shared__ int   parti[128];
  __shared__ float rowmax[16], rowinv[16];
  __shared__ int   rowcnt[16];
  __shared__ float pws[3], thrs[16];

  int i0 = blockIdx.x * 16;
  int bh = blockIdx.y;
  int bb = bh >> 4, hh = bh & 15;
  int tid = threadIdx.x;
  int r = tid >> 3;   // row in tile 0..15
  int g = tid & 7;    // column group 0..7

  if (tid < 3)  pws[tid]  = pw[bb * 3 + tid];
  if (tid < 16) thrs[tid] = thr[(size_t)bh * L_ + i0 + tid];
  __syncthreads();

  const float* srow = sc + ((size_t)bh * L_ + (i0 + r)) * L_;
  float p0 = pws[0], p1 = pws[1], p2 = pws[2];
  float tr = thrs[r];
  int irow = i0 + r;

  // pass 1: masked row max + count
  float mx = -__builtin_inff();
  int cnt = 0;
  for (int j = g; j < L_; j += 8) {
    float s = srow[j];
    int d = irow - j; if (d < 0) d = -d;
    float c = p1 + ((d <= HALFW) ? p0 : 0.f) + ((s >= tr) ? p2 : 0.f);
    if (c > THRESH) { ++cnt; if (s > mx) mx = s; }
  }
  partf[tid] = mx;
  parti[tid] = cnt;
  __syncthreads();
  if (tid < 16) {
    float m = -__builtin_inff();
    int c = 0;
    for (int q = 0; q < 8; ++q) {
      float v = partf[tid * 8 + q];
      if (v > m) m = v;
      c += parti[tid * 8 + q];
    }
    rowcnt[tid] = c;
    rowmax[tid] = (c == 0) ? 0.f : m;
  }
  __syncthreads();
  float rmx = rowmax[r];
  int rcnt = rowcnt[r];

  // pass 2: exponentials into LDS (f16), partial sums
  float sm = 0.f;
  if (rcnt == 0) {
    for (int j = g; j < L_; j += 8) {
      float e = (j == irow) ? 1.f : 0.f;
      ph[r][j] = (f16)e;
      sm += e;
    }
  } else {
    for (int j = g; j < L_; j += 8) {
      float s = srow[j];
      int d = irow - j; if (d < 0) d = -d;
      float c = p1 + ((d <= HALFW) ? p0 : 0.f) + ((s >= tr) ? p2 : 0.f);
      float e = (c > THRESH) ? expf(s - rmx) : 0.f;
      ph[r][j] = (f16)e;
      sm += e;
    }
  }
  partf[tid] = sm;
  __syncthreads();
  if (tid < 16) {
    float s = 0.f;
    for (int q = 0; q < 8; ++q) s += partf[tid * 8 + q];
    rowinv[tid] = 1.0f / s;
  }
  __syncthreads();
  float inv = rowinv[r];
  for (int j = g; j < L_; j += 8) ph[r][j] = (f16)((float)ph[r][j] * inv);
  __syncthreads();

  // WMMA phase: wave w computes the 16x16 d-subtile w of the 16x64 output.
  // V is staged 128 rows at a time into LDS with async global->LDS copies:
  // each of the 128 threads issues 8x 16B copies (VDST = per-lane LDS byte
  // offset = low 32 bits of the generic shared pointer).
  int w = tid >> 5;
  v8f acc = {};
  unsigned ldsbase = (unsigned)(uintptr_t)&vtile[0];
  const f16* gvb = vh + (size_t)bh * L_ * HD_;
  for (int j0 = 0; j0 < L_; j0 += 128) {
    const char* gsrc = (const char*)(gvb + (size_t)j0 * HD_);
#pragma unroll
    for (int t = 0; t < 8; ++t) {
      unsigned loff = ldsbase + (unsigned)((t * 128 + tid) * 16);
      unsigned long long ga =
          (unsigned long long)(uintptr_t)(gsrc + (size_t)(t * 128 + tid) * 16);
      asm volatile("global_load_async_to_lds_b128 %0, %1, off"
                   :: "v"(loff), "v"(ga) : "memory");
    }
    asm volatile("s_wait_asynccnt 0x0" ::: "memory");
    __syncthreads();
#pragma unroll
    for (int jj = 0; jj < 128; jj += 32) {
      v16h a = load_a_frag(&ph[0][j0 + jj], 1024);        // LDS A tile
      v16h b = load_b_frag_k(&vtile[jj * 64 + w * 16], 64);  // LDS B tile
      acc = wmma16(a, b, acc);
    }
    __syncthreads();
  }

  int lane = tid & 31;
  int n = lane & 15, mb = (lane & 16) ? 8 : 0;
  f16* obase = ao + (size_t)bb * L_ * D_ + hh * 64 + w * 16;
#pragma unroll
  for (int v = 0; v < 8; ++v) {
    int i = i0 + mb + v;
    obase[(size_t)i * D_ + n] = (f16)acc[v];
  }
}

// ---------------------------------------------------------------------------
// out = attn_out @ w_proj^T + b_proj   ([4096x1024] f16 x f16 -> f32)
// ---------------------------------------------------------------------------
__global__ void k_gemm_proj(const f16* __restrict__ ah, const f16* __restrict__ wph,
                            const float* __restrict__ bias, float* __restrict__ out) {
  const int NT = 16;  // 1024 / 64
  int wave = blockIdx.x * 8 + (threadIdx.x >> 5);
  int mt = wave / NT, nt = wave % NT;
  int m0 = mt * 16, n0 = nt * 64;
  v8f a0 = {}, a1 = {}, a2 = {}, a3 = {};
  for (int k0 = 0; k0 < D_; k0 += 32) {
    __builtin_prefetch(ah + m0 * D_ + k0 + 64, 0, 1);
    v16h a  = load_a_frag(ah + (size_t)m0 * D_ + k0, D_);
    v16h b0 = load_b_frag_n(wph + (size_t)(n0 +  0) * D_ + k0, D_);
    v16h b1 = load_b_frag_n(wph + (size_t)(n0 + 16) * D_ + k0, D_);
    v16h b2 = load_b_frag_n(wph + (size_t)(n0 + 32) * D_ + k0, D_);
    v16h b3 = load_b_frag_n(wph + (size_t)(n0 + 48) * D_ + k0, D_);
    a0 = wmma16(a, b0, a0);
    a1 = wmma16(a, b1, a1);
    a2 = wmma16(a, b2, a2);
    a3 = wmma16(a, b3, a3);
  }
  int lane = threadIdx.x & 31;
  int n = lane & 15, mb = (lane & 16) ? 8 : 0;
#pragma unroll
  for (int v = 0; v < 8; ++v) {
    float* row = out + (size_t)(m0 + mb + v) * D_ + n0;
    row[ 0 + n] = a0[v] + bias[n0 +  0 + n];
    row[16 + n] = a1[v] + bias[n0 + 16 + n];
    row[32 + n] = a2[v] + bias[n0 + 32 + n];
    row[48 + n] = a3[v] + bias[n0 + 48 + n];
  }
}

// ---------------------------------------------------------------------------
extern "C" void kernel_launch(void* const* d_in, const int* in_sizes, int n_in,
                              void* d_out, int out_size, void* d_ws, size_t ws_size,
                              hipStream_t stream) {
  const float* x      = (const float*)d_in[0];
  const float* w_qkv  = (const float*)d_in[1];
  const float* w_proj = (const float*)d_in[2];
  const float* b_proj = (const float*)d_in[3];
  const float* ps_w1  = (const float*)d_in[4];
  const float* ps_b1  = (const float*)d_in[5];
  const float* ln_g   = (const float*)d_in[6];
  const float* ln_b   = (const float*)d_in[7];
  const float* ps_w2  = (const float*)d_in[8];
  const float* ps_b2  = (const float*)d_in[9];
  const float* ps_w3  = (const float*)d_in[10];
  const float* ps_b3  = (const float*)d_in[11];
  const float* pbias  = (const float*)d_in[12];
  float* out = (float*)d_out;

  // workspace carve (aligned to 256B)
  char* w = (char*)d_ws;
  auto carve = [&](size_t bytes) {
    char* p = w;
    w += (bytes + 255) & ~(size_t)255;
    return p;
  };
  f16* xh     = (f16*)carve((size_t)4096 * 1024 * 2);
  f16* wqkvh  = (f16*)carve((size_t)3072 * 1024 * 2);
  f16* wprojh = (f16*)carve((size_t)1024 * 1024 * 2);
  f16* qh     = (f16*)carve((size_t)B_ * H_ * L_ * HD_ * 2);
  f16* kh     = (f16*)carve((size_t)B_ * H_ * L_ * HD_ * 2);
  f16* vh     = (f16*)carve((size_t)B_ * H_ * L_ * HD_ * 2);
  float* pooled = (float*)carve((size_t)B_ * D_ * 4);
  float* h1     = (float*)carve((size_t)B_ * D_ * 4);
  float* h2     = (float*)carve((size_t)B_ * 512 * 4);
  float* pw     = (float*)carve(64);
  float* thr    = (float*)carve((size_t)B_ * H_ * L_ * 4);
  f16*   ao     = (f16*)carve((size_t)4096 * 1024 * 2);
  float* sc     = (float*)carve((size_t)B_ * H_ * L_ * L_ * 4);  // 256 MiB

  // 1) f16 conversions
  k_cvt<<<(4096 * 1024 + 255) / 256, 256, 0, stream>>>(x, xh, 4096 * 1024);
  k_cvt<<<(3072 * 1024 + 255) / 256, 256, 0, stream>>>(w_qkv, wqkvh, 3072 * 1024);
  k_cvt<<<(1024 * 1024 + 255) / 256, 256, 0, stream>>>(w_proj, wprojh, 1024 * 1024);

  // 2) QKV projection (WMMA): 256*48 = 12288 waves
  k_gemm_qkv<<<1536, 256, 0, stream>>>(xh, wqkvh, qh, kh, vh);

  // 3) pattern selector MLP
  k_pool<<<(B_ * D_ + 255) / 256, 256, 0, stream>>>(x, pooled);
  k_dot<<<512, 256, 0, stream>>>(pooled, ps_w1, ps_b1, h1, D_, D_, 0);     // 4096 waves
  k_ln_gelu<<<B_, 256, 0, stream>>>(h1, ln_g, ln_b);
  k_dot<<<256, 256, 0, stream>>>(h1, ps_w2, ps_b2, h2, D_, 512, 1);        // 2048 waves
  k_logits<<<B_, 96, 0, stream>>>(h2, ps_w3, ps_b3, pbias, pw);

  // 4) scores (WMMA): per bh, 1024 waves
  k_scores<<<dim3(128, B_ * H_), 256, 0, stream>>>(qh, kh, sc);

  // 5) per-row exact K_TOP-th largest (sparse_w/sparse_b are a positive
  //    affine per head -> top-k indices of s2 == top-k indices of scores)
  k_thresh<<<(B_ * H_ * L_) / 8, 256, 0, stream>>>(sc, thr);

  // 6) fused mask + softmax + P@V (WMMA + async global->LDS staging)
  k_attn<<<dim3(L_ / 16, B_ * H_), 128, 0, stream>>>(sc, thr, pw, vh, ao);

  // 7) output projection (WMMA) + bias -> f32 d_out
  k_gemm_proj<<<512, 256, 0, stream>>>(ao, wprojh, b_proj, out);

  (void)in_sizes; (void)n_in; (void)out_size; (void)ws_size;
}
